// GRUAggregation1d_37091337568325
// MI455X (gfx1250) — matrix operand order repaired and checked by reading, not compiled, not run on hardware
//
#include <hip/hip_runtime.h>

typedef __bf16 bf16;
typedef __attribute__((ext_vector_type(16))) __bf16 v16bf;
typedef __attribute__((ext_vector_type(8)))  __bf16 v8bf;
typedef __attribute__((ext_vector_type(8)))  float  v8f;
typedef __attribute__((ext_vector_type(4)))  float  f32x4;

#define WIDTH   128
#define IN_DIM  512
#define BATCH   16
#define LSEQ    8192
#define TILE_L  64
#define NCHUNKS (IN_DIM / 32)   // 16
#define XS_PITCH 40    // bf16 elems per column (32 used); 80B row stride -> 16B aligned
#define Z_PITCH  136   // bf16 elems per column (128 used); 272B row stride -> 16B aligned

__device__ __forceinline__ v8f wmma_bf16(v16bf a, v16bf b, v8f c) {
    return __builtin_amdgcn_wmma_f32_16x16x32_bf16(false, a, false, b, (short)0, c, false, false);
}

// A fragment (16x32 bf16, this lane's slice) from an fp32 weight matrix, converting on the fly.
// lane%16 = row within M-tile; lane/16 selects K range {0-15} vs {16-31}.
__device__ __forceinline__ v16bf load_a_frag(const float* __restrict__ W, int row, int ld, int k) {
    const float* p = W + (size_t)row * ld + k;
    f32x4 f0 = *(const f32x4*)(p);
    f32x4 f1 = *(const f32x4*)(p + 4);
    f32x4 g0 = *(const f32x4*)(p + 8);
    f32x4 g1 = *(const f32x4*)(p + 12);
    v16bf a;
    a[0]=(bf16)f0.x;  a[1]=(bf16)f0.y;  a[2]=(bf16)f0.z;  a[3]=(bf16)f0.w;
    a[4]=(bf16)f1.x;  a[5]=(bf16)f1.y;  a[6]=(bf16)f1.z;  a[7]=(bf16)f1.w;
    a[8]=(bf16)g0.x;  a[9]=(bf16)g0.y;  a[10]=(bf16)g0.z; a[11]=(bf16)g0.w;
    a[12]=(bf16)g1.x; a[13]=(bf16)g1.y; a[14]=(bf16)g1.z; a[15]=(bf16)g1.w;
    return a;
}

// B fragment: 16 contiguous bf16 K-values of this lane's column, from LDS (two 16B loads).
__device__ __forceinline__ v16bf load_b_frag(const bf16* __restrict__ p) {
    v8bf lo = *(const v8bf*)(p);
    v8bf hi = *(const v8bf*)(p + 8);
    v16bf bm;
#pragma unroll
    for (int i = 0; i < 8; ++i) { bm[i] = lo[i]; bm[8 + i] = hi[i]; }
    return bm;
}

// xs chunk staging: thread owns (col, 8 contiguous k). Fully unrolled coalesced loads.
__device__ __forceinline__ void xs_load_regs(const float* __restrict__ xs, size_t xs_base,
                                             int kc, int col, int k0, float xf[8]) {
#pragma unroll
    for (int i = 0; i < 8; ++i)
        xf[i] = xs[xs_base + (size_t)(kc * 32 + k0 + i) * LSEQ + col];
}

// Pack 8 fp32 -> 8 bf16 (v_cvt_pk_bf16_f32 x4) and store as one 16B ds_store_b128.
__device__ __forceinline__ void xs_store_lds(bf16* __restrict__ dst, int col, int k0,
                                             const float xf[8]) {
    v8bf v;
#pragma unroll
    for (int i = 0; i < 8; ++i) v[i] = (bf16)xf[i];
    *(v8bf*)(dst + col * XS_PITCH + k0) = v;
}

__global__ __launch_bounds__(256)
void gru_fused_wmma(const float* __restrict__ xs, const float* __restrict__ zp,
                    const float* __restrict__ Wq, const float* __restrict__ Wq_b,
                    const float* __restrict__ Uq, const float* __restrict__ Uq_b,
                    const float* __restrict__ Wr, const float* __restrict__ Wr_b,
                    const float* __restrict__ Ur, const float* __restrict__ Ur_b,
                    const float* __restrict__ Wz, const float* __restrict__ Wz_b,
                    const float* __restrict__ Uz, const float* __restrict__ Uz_b,
                    float* __restrict__ out)
{
    __shared__ bf16 lds_xs[2][TILE_L * XS_PITCH];  // 2 x 5.0 KB (double buffer)
    __shared__ bf16 lds_z [TILE_L * Z_PITCH];      // 17.0 KB
    __shared__ bf16 lds_rz[TILE_L * Z_PITCH];      // 17.0 KB

    const int tid  = threadIdx.x;
    const int lane = tid & 31;
    const int wv   = tid >> 5;           // 0..7 : wave owns M rows [16*wv, 16*wv+16)
    const int m0   = wv * 16;
    const int kh   = lane >> 4;          // 0/1 : K-half selector
    const int l16  = lane & 15;

    const int bid = blockIdx.x;
    const int b   = bid >> 7;            // / (LSEQ / TILE_L)
    const int l0  = (bid & 127) * TILE_L;

    const size_t xs_base = ((size_t)b * IN_DIM) * LSEQ + l0;
    const size_t z_base  = ((size_t)b * WIDTH)  * LSEQ + l0;

    // staging ownership: one column, contiguous K run
    const int scol = tid & 63;
    const int sk0  = (tid >> 6) * 8;     // xs: 8 contiguous k per thread

    // ---- Stage z_prev tile (128 x 64) as bf16, column-major [col][k] ----
    {
        const int zk0 = (tid >> 6) * 32; // z: 32 contiguous k per thread
        float zf[32];
#pragma unroll
        for (int k = 0; k < 32; ++k)
            zf[k] = zp[z_base + (size_t)(zk0 + k) * LSEQ + scol];
#pragma unroll
        for (int j = 0; j < 4; ++j) {
            v8bf v;
#pragma unroll
            for (int i = 0; i < 8; ++i) v[i] = (bf16)zf[8 * j + i];
            *(v8bf*)&lds_z[scol * Z_PITCH + zk0 + 8 * j] = v;
        }
    }

    // Per-lane combined biases for this lane's 8 accumulator rows
    float biasQ[8], biasR[8], biasZ[8];
#pragma unroll
    for (int j = 0; j < 8; ++j) {
        int row = m0 + j + 8 * kh;
        biasQ[j] = Wq_b[row] + Uq_b[row];
        biasR[j] = Wr_b[row] + Ur_b[row];
        biasZ[j] = Wz_b[row] + Uz_b[row];
    }

    v8f accQ[4] = {}, accR[4] = {}, accZ[4] = {};

    const int arow = m0 + l16;           // A-fragment row for this lane
    const int akb  = kh * 16;            // K sub-base inside a 32-chunk

    // ---- prologue: stage xs chunk 0 into buffer 0 ----
    float xf[8];
    xs_load_regs(xs, xs_base, 0, scol, sk0, xf);
    xs_store_lds(lds_xs[0], scol, sk0, xf);
    __syncthreads();   // also covers z tile visibility

    // ---- W projections: K = IN_DIM in chunks of 32, double-buffered ----
    for (int kc = 0; kc < NCHUNKS; ++kc) {
        const int cur  = kc & 1;
        const bool more = (kc + 1) < NCHUNKS;

        // kick off next chunk's global loads early (hidden behind WMMAs)
        if (more) xs_load_regs(xs, xs_base, kc + 1, scol, sk0, xf);

        // prefetch two chunks ahead (clamped address, no branch)
        const int kp = (kc + 2 < NCHUNKS) ? kc + 2 : NCHUNKS - 1;
        __builtin_prefetch(&xs[xs_base + (size_t)(kp * 32 + sk0) * LSEQ + scol], 0, 1);

        const int k0 = kc * 32;
        v16bf aq = load_a_frag(Wq, arow, IN_DIM, k0 + akb);
        v16bf ar = load_a_frag(Wr, arow, IN_DIM, k0 + akb);
        v16bf az = load_a_frag(Wz, arow, IN_DIM, k0 + akb);
#pragma unroll
        for (int nt = 0; nt < 4; ++nt) {
            v16bf bm = load_b_frag(&lds_xs[cur][(nt * 16 + l16) * XS_PITCH + kh * 16]);
            accQ[nt] = wmma_bf16(aq, bm, accQ[nt]);
            accR[nt] = wmma_bf16(ar, bm, accR[nt]);
            accZ[nt] = wmma_bf16(az, bm, accZ[nt]);
        }

        if (more) xs_store_lds(lds_xs[cur ^ 1], scol, sk0, xf);
        __syncthreads();
    }

    // ---- Uq / Ur projections: K = WIDTH in chunks of 32 (B from staged z) ----
#pragma unroll
    for (int kc = 0; kc < WIDTH / 32; ++kc) {
        const int k0 = kc * 32;
        v16bf auq = load_a_frag(Uq, arow, WIDTH, k0 + akb);
        v16bf aur = load_a_frag(Ur, arow, WIDTH, k0 + akb);
#pragma unroll
        for (int nt = 0; nt < 4; ++nt) {
            v16bf bm = load_b_frag(&lds_z[(nt * 16 + l16) * Z_PITCH + k0 + kh * 16]);
            accQ[nt] = wmma_bf16(auq, bm, accQ[nt]);
            accR[nt] = wmma_bf16(aur, bm, accR[nt]);
        }
    }

    // ---- gates: q = sigmoid(.), r = sigmoid(.); write r*z back to LDS as bf16 ----
#pragma unroll
    for (int nt = 0; nt < 4; ++nt) {
        const int col = nt * 16 + l16;
#pragma unroll
        for (int j = 0; j < 8; ++j) {
            const int row = m0 + j + 8 * kh;
            float q = 1.0f / (1.0f + __expf(-(accQ[nt][j] + biasQ[j])));
            float r = 1.0f / (1.0f + __expf(-(accR[nt][j] + biasR[j])));
            float z = zp[z_base + (size_t)row * LSEQ + col];
            lds_rz[col * Z_PITCH + row] = (bf16)(r * z);
            accQ[nt][j] = q;   // keep gate q
            accR[nt][j] = z;   // keep z_prev value for final blend
        }
    }
    __syncthreads();

    // ---- Uz @ (r*z): K = WIDTH in chunks of 32 ----
#pragma unroll
    for (int kc = 0; kc < WIDTH / 32; ++kc) {
        const int k0 = kc * 32;
        v16bf auz = load_a_frag(Uz, arow, WIDTH, k0 + akb);
#pragma unroll
        for (int nt = 0; nt < 4; ++nt) {
            v16bf bm = load_b_frag(&lds_rz[(nt * 16 + l16) * Z_PITCH + k0 + kh * 16]);
            accZ[nt] = wmma_bf16(auz, bm, accZ[nt]);
        }
    }

    // ---- z_tilde = tanh(.); out = q*z + (1-q)*z_tilde ----
#pragma unroll
    for (int nt = 0; nt < 4; ++nt) {
        const int col = nt * 16 + l16;
#pragma unroll
        for (int j = 0; j < 8; ++j) {
            const int row = m0 + j + 8 * kh;
            float e  = __expf(2.0f * (accZ[nt][j] + biasZ[j]));
            float zt = (e - 1.0f) / (e + 1.0f);
            float q  = accQ[nt][j];
            float z  = accR[nt][j];
            out[z_base + (size_t)row * LSEQ + col] = q * z + (1.0f - q) * zt;
        }
    }
}

extern "C" void kernel_launch(void* const* d_in, const int* in_sizes, int n_in,
                              void* d_out, int out_size, void* d_ws, size_t ws_size,
                              hipStream_t stream) {
    const float* xs   = (const float*)d_in[0];
    const float* zp   = (const float*)d_in[1];
    const float* Wq_w = (const float*)d_in[2];
    const float* Wq_b = (const float*)d_in[3];
    const float* Uq_w = (const float*)d_in[4];
    const float* Uq_b = (const float*)d_in[5];
    const float* Wr_w = (const float*)d_in[6];
    const float* Wr_b = (const float*)d_in[7];
    const float* Ur_w = (const float*)d_in[8];
    const float* Ur_b = (const float*)d_in[9];
    const float* Wz_w = (const float*)d_in[10];
    const float* Wz_b = (const float*)d_in[11];
    const float* Uz_w = (const float*)d_in[12];
    const float* Uz_b = (const float*)d_in[13];
    float* out = (float*)d_out;

    dim3 grid(BATCH * (LSEQ / TILE_L));   // 16 * 128 = 2048 blocks
    dim3 block(256);                      // 8 wave32s
    hipLaunchKernelGGL(gru_fused_wmma, grid, block, 0, stream,
                       xs, zp, Wq_w, Wq_b, Uq_w, Uq_b,
                       Wr_w, Wr_b, Ur_w, Ur_b, Wz_w, Wz_b, Uz_w, Uz_b, out);
}
